// EncoderBlock_52493090291723
// MI455X (gfx1250) — matrix operand order repaired and checked by reading, not compiled, not run on hardware
//
#include <hip/hip_runtime.h>
#include <math.h>

// ---------------------------------------------------------------------------
// CDNA5 / gfx1250 encoder block: LN -> QKV -> flash-attn -> Wo(+res) -> LN
//                                -> W1+GELU -> W2(+res)
// Matmuls: v_wmma_f32_16x16x32_bf16 (wave32), fp32 accumulate.
// Tile staging: global_load_async_to_lds_b128 (ASYNCcnt) — CDNA5 async path.
// ---------------------------------------------------------------------------

typedef __attribute__((ext_vector_type(16))) __bf16 v16bf;
typedef __attribute__((ext_vector_type(8)))  float  v8f;

union Frag  { v16bf v; uint4 q[2]; };   // 16 bf16 = 32B = one WMMA A/B operand
union Half8 { uint4  q; __bf16 e[8]; };

static __device__ __forceinline__
v8f wmma_bf16(const Frag& a, const Frag& b, v8f c) {
  return __builtin_amdgcn_wmma_f32_16x16x32_bf16(
      /*neg_a=*/false, a.v, /*neg_b=*/false, b.v,
      /*c_mod=*/(short)0, c, /*reuse_a=*/false, /*reuse_b=*/false);
}

// LDS byte offset of a __shared__ pointer (flat LDS aperture: addr[31:0]).
static __device__ __forceinline__ uint32_t lds_off(const void* p) {
  return (uint32_t)(uintptr_t)p;
}

// Async DMA: 16B per lane, global -> LDS, tracked by ASYNCcnt.
static __device__ __forceinline__
void async_b128(uint32_t lds, const void* gaddr) {
  asm volatile("global_load_async_to_lds_b128 %0, %1, off"
               :: "v"(lds), "v"(gaddr) : "memory");
}
static __device__ __forceinline__
void async_b128_o16(uint32_t lds, const void* gaddr) {  // +16B both sides
  asm volatile("global_load_async_to_lds_b128 %0, %1, off offset:16"
               :: "v"(lds), "v"(gaddr) : "memory");
}
static __device__ __forceinline__ void wait_async0() {
  asm volatile("s_wait_asynccnt 0x0" ::: "memory");
}

// ---------------------------------------------------------------------------
// LayerNorm over D=1024, one 256-thread block per row, bf16 output.
// ---------------------------------------------------------------------------
__global__ __launch_bounds__(256)
void ln_bf16_kernel(const float* __restrict__ x, const float* __restrict__ g,
                    const float* __restrict__ bt, __bf16* __restrict__ out) {
  const int D = 1024;
  const int row = blockIdx.x;
  const float* xr = x + (size_t)row * D;
  __shared__ float s_sum[8], s_sq[8];
  const int t = threadIdx.x;
  float v[4], sum = 0.f, sq = 0.f;
  #pragma unroll
  for (int i = 0; i < 4; ++i) {
    float f = xr[t + 256 * i];
    v[i] = f; sum += f; sq += f * f;
  }
  #pragma unroll
  for (int m = 16; m >= 1; m >>= 1) {
    sum += __shfl_xor(sum, m);
    sq  += __shfl_xor(sq,  m);
  }
  if ((t & 31) == 0) { s_sum[t >> 5] = sum; s_sq[t >> 5] = sq; }
  __syncthreads();
  float tot = 0.f, tot2 = 0.f;
  #pragma unroll
  for (int i = 0; i < 8; ++i) { tot += s_sum[i]; tot2 += s_sq[i]; }
  const float mu = tot * (1.f / 1024.f);
  const float var = tot2 * (1.f / 1024.f) - mu * mu;
  const float rs = rsqrtf(var + 1e-5f);
  __bf16* orow = out + (size_t)row * D;
  #pragma unroll
  for (int i = 0; i < 4; ++i) {
    int c = t + 256 * i;
    orow[c] = (__bf16)((v[i] - mu) * rs * g[c] + bt[c]);
  }
}

// ---------------------------------------------------------------------------
// Weight conversion: fp32 -> bf16, stored TRANSPOSED [N x K].
// ---------------------------------------------------------------------------
__global__ __launch_bounds__(256)
void cvt_qkv_t_kernel(const float* __restrict__ W /*[H,D,HD]*/,
                      __bf16* __restrict__ out /*[N=D][K=D]*/) {
  int idx = blockIdx.x * 256 + threadIdx.x;
  int d = idx & 1023;
  int n = idx >> 10;
  int h = n >> 6, e = n & 63;
  out[(size_t)n * 1024 + d] = (__bf16)W[((size_t)h * 1024 + d) * 64 + e];
}

__global__ __launch_bounds__(256)
void cvt_t_kernel(const float* __restrict__ W /*[K][N]*/,
                  __bf16* __restrict__ out /*[N][K]*/) {
  int idx = blockIdx.x * 256 + threadIdx.x;
  int n = idx & 1023;
  int k = idx >> 10;
  out[(size_t)n * 1024 + k] = (__bf16)W[(size_t)k * 1024 + n];
}

// ---------------------------------------------------------------------------
// GEMM: C[M,N] = A[M,K](bf16) * Bt[N,K]^T(bf16)  (+bias)(+GELU)(+residual)
// 128x128 block tile, K-step 32, double-buffered LDS via async DMA,
// 8 waves (2Mx4N), each wave 4x2 16x16 tiles -> 8 WMMAs / K-step.
// LDS row stride padded to 40 elems (80B = 20 banks): conflict-free b128.
// ---------------------------------------------------------------------------
#define TM 128
#define TN 128
#define TK 32
#define LDA 40   // padded LDS row stride (elements)

__global__ __launch_bounds__(256)
void gemm_bf16_kernel(const __bf16* __restrict__ A,
                      const __bf16* __restrict__ Bt,
                      const float* __restrict__ bias,
                      const float* __restrict__ resid,
                      float* __restrict__ Cf,
                      __bf16* __restrict__ Cb,
                      int M, int N, int K, int gelu) {
  __shared__ __bf16 As[2][TM * LDA];
  __shared__ __bf16 Bs[2][TN * LDA];

  const int tid  = threadIdx.x;
  const int lane = tid & 31;
  const int wave = tid >> 5;
  const int wm = wave & 1;        // 0..1  (64 rows each)
  const int wn = wave >> 1;       // 0..3  (32 cols each)
  const int l16 = lane & 15;
  const int hi  = lane >> 4;

  const int mrow = blockIdx.x * TM;
  const int ncol = blockIdx.y * TN;

  // Cooperative staging: 256 threads x 32B (two async b128) per matrix.
  const int lr = tid >> 1;              // 0..127
  const int lc = (tid & 1) * 16;        // 0 | 16
  const __bf16* Ag = A  + (size_t)(mrow + lr) * K + lc;
  const __bf16* Bg = Bt + (size_t)(ncol + lr) * K + lc;
  const uint32_t ldsA0 = lds_off(&As[0][lr * LDA + lc]);
  const uint32_t ldsB0 = lds_off(&Bs[0][lr * LDA + lc]);
  const uint32_t BUFB  = (uint32_t)(TM * LDA * sizeof(__bf16));

  auto stage = [&](int kb, int buf) {
    const __bf16* ap = Ag + kb * TK;
    const __bf16* bp = Bg + kb * TK;
    const uint32_t la = ldsA0 + buf * BUFB;
    const uint32_t lb = ldsB0 + buf * BUFB;
    async_b128(la, ap);  async_b128_o16(la, ap);
    async_b128(lb, bp);  async_b128_o16(lb, bp);
  };

  const v8f zero8 = {0.f, 0.f, 0.f, 0.f, 0.f, 0.f, 0.f, 0.f};
  v8f acc[4][2];
  #pragma unroll
  for (int mt = 0; mt < 4; ++mt)
    #pragma unroll
    for (int nt = 0; nt < 2; ++nt) acc[mt][nt] = zero8;

  const int nk = K / TK;
  stage(0, 0);
  wait_async0();
  __syncthreads();

  for (int kb = 0; kb < nk; ++kb) {
    const int buf = kb & 1;
    if (kb + 1 < nk) {
      stage(kb + 1, buf ^ 1);            // DMA overlaps the WMMAs below
      if (kb + 2 < nk) {
        __builtin_prefetch(Ag + (kb + 2) * TK, 0, 1);
        __builtin_prefetch(Bg + (kb + 2) * TK, 0, 1);
      }
    }
    Frag af[4];
    #pragma unroll
    for (int mt = 0; mt < 4; ++mt) {
      const __bf16* p = &As[buf][(wm * 64 + mt * 16 + l16) * LDA];
      af[mt].q[0] = *(const uint4*)(p + hi * 16);
      af[mt].q[1] = *(const uint4*)(p + hi * 16 + 8);
    }
    Frag bf[2];
    #pragma unroll
    for (int nt = 0; nt < 2; ++nt) {
      const __bf16* p = &Bs[buf][(wn * 32 + nt * 16 + l16) * LDA + hi * 16];
      bf[nt].q[0] = *(const uint4*)(p);
      bf[nt].q[1] = *(const uint4*)(p + 8);
    }
    #pragma unroll
    for (int mt = 0; mt < 4; ++mt)
      #pragma unroll
      for (int nt = 0; nt < 2; ++nt)
        acc[mt][nt] = wmma_bf16(af[mt], bf[nt], acc[mt][nt]);
    wait_async0();                       // publish next buffer
    __syncthreads();
  }

  // Epilogue (C layout: VGPR r -> row r + 8*hi, lane l16 -> col)
  #pragma unroll
  for (int mt = 0; mt < 4; ++mt) {
    #pragma unroll
    for (int nt = 0; nt < 2; ++nt) {
      const int gc = ncol + wn * 32 + nt * 16 + l16;
      #pragma unroll
      for (int r = 0; r < 8; ++r) {
        const int gr = mrow + wm * 64 + mt * 16 + hi * 8 + r;
        float c = acc[mt][nt][r];
        if (bias)  c += bias[gc];
        if (gelu)  c = 0.5f * c * (1.f + erff(c * 0.70710678118654752f));
        if (resid) c += resid[(size_t)gr * N + gc];
        if (Cf) Cf[(size_t)gr * N + gc] = c;
        if (Cb) Cb[(size_t)gr * N + gc] = (__bf16)c;
      }
    }
  }
}

// ---------------------------------------------------------------------------
// Fused flash attention.  Block = (b, h, 128 q-rows), 8 waves, each wave owns
// one 16-row Q tile in registers.  Streams 32-key chunks of K (async DMA) and
// V (transposed) through LDS; online softmax; P via per-wave LDS scratch.
// Padded LDS strides: K 72 (144B), Vt/P 40 (80B) -> conflict-free b128 reads.
// ---------------------------------------------------------------------------
#define KSP 72
#define VTP 40
#define PSP 40

__global__ __launch_bounds__(256)
void attn_kernel(const __bf16* __restrict__ Q, const __bf16* __restrict__ Km,
                 const __bf16* __restrict__ Vm, __bf16* __restrict__ O) {
  const int S = 1024, D = 1024, HD = 64;
  const int qblk = blockIdx.x;
  const int h    = blockIdx.y;
  const int b    = blockIdx.z;
  const int tid  = threadIdx.x;
  const int lane = tid & 31;
  const int wave = tid >> 5;
  const int l16  = lane & 15;
  const int hi   = lane >> 4;

  __shared__ __bf16 Ks[32 * KSP];       // [key][d], padded
  __shared__ __bf16 Vt[64 * VTP];       // [e][key], padded
  __shared__ __bf16 Ps[8][16 * PSP];    // per-wave P scratch [row][key]

  // Q tile in registers: two A fragments covering d 0..31 and 32..63
  const int qrow = qblk * 128 + wave * 16 + l16;
  const __bf16* qp = Q + (size_t)(b * S + qrow) * D + h * HD;
  Frag qf[2];
  qf[0].q[0] = *(const uint4*)(qp + hi * 16);
  qf[0].q[1] = *(const uint4*)(qp + hi * 16 + 8);
  qf[1].q[0] = *(const uint4*)(qp + 32 + hi * 16);
  qf[1].q[1] = *(const uint4*)(qp + 32 + hi * 16 + 8);

  const v8f zero8 = {0.f, 0.f, 0.f, 0.f, 0.f, 0.f, 0.f, 0.f};
  v8f o[4];
  #pragma unroll
  for (int et = 0; et < 4; ++et) o[et] = zero8;
  float mrow[8], lrow[8];
  #pragma unroll
  for (int r = 0; r < 8; ++r) { mrow[r] = -1e30f; lrow[r] = 0.f; }

  const int key = tid >> 3;             // 0..31 (staging role)
  const int seg = (tid & 7) * 8;        // 0..56 (staging role)
  const uint32_t ldsK = lds_off(&Ks[key * KSP + seg]);

  for (int c = 0; c < S / 32; ++c) {
    __syncthreads();                    // prior reads of Ks/Vt done
    {
      const size_t kvrow = (size_t)(b * S + c * 32 + key) * D + h * HD + seg;
      async_b128(ldsK, Km + kvrow);     // K chunk: straight async DMA
      Half8 vv; vv.q = *(const uint4*)(Vm + kvrow);   // V chunk: transpose in
      #pragma unroll
      for (int j = 0; j < 8; ++j) Vt[(seg + j) * VTP + key] = vv.e[j];
    }
    wait_async0();
    __syncthreads();

    // scores: S16x32 = Q(16x64) * K^T(64x32)
    v8f s[2];
    #pragma unroll
    for (int j = 0; j < 2; ++j) {
      Frag kf0, kf1;
      const __bf16* p0 = &Ks[(j * 16 + l16) * KSP + hi * 16];
      kf0.q[0] = *(const uint4*)(p0);
      kf0.q[1] = *(const uint4*)(p0 + 8);
      const __bf16* p1 = p0 + 32;
      kf1.q[0] = *(const uint4*)(p1);
      kf1.q[1] = *(const uint4*)(p1 + 8);
      s[j] = wmma_bf16(qf[0], kf0, zero8);
      s[j] = wmma_bf16(qf[1], kf1, s[j]);
    }

    // online softmax (rows live across 16 lanes of a half-wave)
    float alpha[8];
    #pragma unroll
    for (int r = 0; r < 8; ++r) {
      float a0 = s[0][r] * 0.125f;      // HD^-0.5
      float a1 = s[1][r] * 0.125f;
      float mx = fmaxf(a0, a1);
      #pragma unroll
      for (int msk = 1; msk < 16; msk <<= 1) mx = fmaxf(mx, __shfl_xor(mx, msk));
      const float mn = fmaxf(mrow[r], mx);
      alpha[r] = __expf(mrow[r] - mn);
      const float p0 = __expf(a0 - mn);
      const float p1 = __expf(a1 - mn);
      s[0][r] = p0; s[1][r] = p1;
      float ps = p0 + p1;
      #pragma unroll
      for (int msk = 1; msk < 16; msk <<= 1) ps += __shfl_xor(ps, msk);
      lrow[r] = lrow[r] * alpha[r] + ps;
      mrow[r] = mn;
    }

    // rescale O; spill P (C layout) -> LDS for A-layout reload
    #pragma unroll
    for (int et = 0; et < 4; ++et)
      #pragma unroll
      for (int r = 0; r < 8; ++r) o[et][r] *= alpha[r];
    #pragma unroll
    for (int j = 0; j < 2; ++j)
      #pragma unroll
      for (int r = 0; r < 8; ++r)
        Ps[wave][(hi * 8 + r) * PSP + j * 16 + l16] = (__bf16)s[j][r];
    __syncthreads();

    Frag pf;
    const __bf16* pp = &Ps[wave][l16 * PSP];
  pf.q[0] = *(const uint4*)(pp + hi * 16);
  pf.q[1] = *(const uint4*)(pp + hi * 16 + 8);

    // O += P(16x32) * V(32x64)
    #pragma unroll
    for (int et = 0; et < 4; ++et) {
      Frag vf;
      const __bf16* vp = &Vt[(et * 16 + l16) * VTP + hi * 16];
      vf.q[0] = *(const uint4*)(vp);
      vf.q[1] = *(const uint4*)(vp + 8);
      o[et] = wmma_bf16(pf, vf, o[et]);
    }
  }

  // normalize and store (bf16, concat-head layout)
  #pragma unroll
  for (int et = 0; et < 4; ++et) {
    #pragma unroll
    for (int r = 0; r < 8; ++r) {
      const int row = qblk * 128 + wave * 16 + hi * 8 + r;
      O[(size_t)(b * S + row) * D + h * HD + et * 16 + l16] =
          (__bf16)(o[et][r] / lrow[r]);
    }
  }
}

// ---------------------------------------------------------------------------
// Launch
// ---------------------------------------------------------------------------
extern "C" void kernel_launch(void* const* d_in, const int* in_sizes, int n_in,
                              void* d_out, int out_size, void* d_ws, size_t ws_size,
                              hipStream_t stream) {
  (void)in_sizes; (void)n_in; (void)out_size; (void)ws_size;
  const int Bn = 8, S = 1024, Dm = 1024, H = 16;
  const int M = Bn * S;                             // 8192

  const float* x    = (const float*)d_in[0];
  const float* Wq   = (const float*)d_in[1];
  const float* Wk   = (const float*)d_in[2];
  const float* Wv   = (const float*)d_in[3];
  const float* Wo   = (const float*)d_in[4];
  const float* bo   = (const float*)d_in[5];
  const float* W1   = (const float*)d_in[6];
  const float* b1   = (const float*)d_in[7];
  const float* W2   = (const float*)d_in[8];
  const float* b2   = (const float*)d_in[9];
  const float* gln1 = (const float*)d_in[10];
  const float* bln1 = (const float*)d_in[11];
  const float* gln2 = (const float*)d_in[12];
  const float* bln2 = (const float*)d_in[13];
  float* out = (float*)d_out;

  char* ws = (char*)d_ws;
  const size_t MB = 1024ull * 1024ull;
  __bf16* Wq_t  = (__bf16*)(ws + 0 * MB);           // 2MB each (D*D bf16)
  __bf16* Wk_t  = (__bf16*)(ws + 2 * MB);
  __bf16* Wv_t  = (__bf16*)(ws + 4 * MB);
  __bf16* Wo_t  = (__bf16*)(ws + 6 * MB);
  __bf16* W1_t  = (__bf16*)(ws + 8 * MB);
  __bf16* W2_t  = (__bf16*)(ws + 10 * MB);
  __bf16* h_b   = (__bf16*)(ws + 12 * MB);          // 16MB (M*D bf16)
  __bf16* Qb    = (__bf16*)(ws + 28 * MB);
  __bf16* Kb    = (__bf16*)(ws + 44 * MB);
  __bf16* Vb    = (__bf16*)(ws + 60 * MB);
  __bf16* attnb = (__bf16*)(ws + 76 * MB);
  float*  x2    = (float*)(ws + 92 * MB);           // 32MB (M*D f32)
  __bf16* h2_b  = (__bf16*)(ws + 124 * MB);
  __bf16* ff1_b = (__bf16*)(ws + 140 * MB);         // end 156MB

  const dim3 blk(256);
  const dim3 gCvt(4096);
  const dim3 gLn(M);
  const dim3 gGemm(M / TM, Dm / TN);                 // 64 x 8
  const dim3 gAttn(S / 128, H, Bn);                  // 8 x 16 x 8

  cvt_qkv_t_kernel<<<gCvt, blk, 0, stream>>>(Wq, Wq_t);
  cvt_qkv_t_kernel<<<gCvt, blk, 0, stream>>>(Wk, Wk_t);
  cvt_qkv_t_kernel<<<gCvt, blk, 0, stream>>>(Wv, Wv_t);
  cvt_t_kernel<<<gCvt, blk, 0, stream>>>(Wo, Wo_t);
  cvt_t_kernel<<<gCvt, blk, 0, stream>>>(W1, W1_t);
  cvt_t_kernel<<<gCvt, blk, 0, stream>>>(W2, W2_t);

  ln_bf16_kernel<<<gLn, blk, 0, stream>>>(x, gln1, bln1, h_b);

  gemm_bf16_kernel<<<gGemm, blk, 0, stream>>>(h_b, Wq_t, nullptr, nullptr,
                                              nullptr, Qb, M, Dm, Dm, 0);
  gemm_bf16_kernel<<<gGemm, blk, 0, stream>>>(h_b, Wk_t, nullptr, nullptr,
                                              nullptr, Kb, M, Dm, Dm, 0);
  gemm_bf16_kernel<<<gGemm, blk, 0, stream>>>(h_b, Wv_t, nullptr, nullptr,
                                              nullptr, Vb, M, Dm, Dm, 0);

  attn_kernel<<<gAttn, blk, 0, stream>>>(Qb, Kb, Vb, attnb);

  gemm_bf16_kernel<<<gGemm, blk, 0, stream>>>(attnb, Wo_t, bo, x,
                                              x2, nullptr, M, Dm, Dm, 0);

  ln_bf16_kernel<<<gLn, blk, 0, stream>>>(x2, gln2, bln2, h2_b);

  gemm_bf16_kernel<<<gGemm, blk, 0, stream>>>(h2_b, W1_t, b1, nullptr,
                                              nullptr, ff1_b, M, Dm, Dm, 1);
  gemm_bf16_kernel<<<gGemm, blk, 0, stream>>>(ff1_b, W2_t, b2, x2,
                                              out, nullptr, M, Dm, Dm, 0);
}
